// ImageZoomxModel_77223511982469
// MI455X (gfx1250) — matrix operand order repaired and checked by reading, not compiled, not run on hardware
//
#include <hip/hip_runtime.h>

// ---------------------------------------------------------------------------
// ImageZoomx (EDSR-16 backbone + LIIF MLP) for MI455X / gfx1250, wave32 WMMA.
// GEMM work on v_wmma_f32_16x16x32_f16; block-shared weights staged into LDS
// via the Tensor Data Mover (tensor_load_to_lds + s_wait_tensorcnt).
// ---------------------------------------------------------------------------

typedef _Float16 h8_t  __attribute__((ext_vector_type(8)));
typedef _Float16 h16_t __attribute__((ext_vector_type(16)));
typedef float    f8_t  __attribute__((ext_vector_type(8)));
typedef unsigned int u32x4 __attribute__((ext_vector_type(4)));
typedef unsigned int u32x8 __attribute__((ext_vector_type(8)));

#define HW        16384      // 128*128 feature pixels
#define N_OUT     147456     // 384*384 query points
#define CONVW_PER 36864      // 9*64*64 halves per conv layer
#define MLP_ACT_BYTES 67584  // 4 waves * 2 ping-pong bufs * 16*264 halves
#define MLP_W_BYTES   131072 // one 256x256 f16 weight matrix

__device__ inline h16_t cat8(h8_t lo, h8_t hi) {
  return __builtin_shufflevector(lo, hi, 0,1,2,3,4,5,6,7,8,9,10,11,12,13,14,15);
}
// A-operand fragment: elements 0..7 at p, elements 8..15 at p+16
// (ISA 16-bit A layout; caller folds the lane>=16 "+8" into p).
__device__ inline h16_t ldA16(const _Float16* p) {
  h8_t lo = *(const h8_t*)p;
  h8_t hi = *(const h8_t*)(p + 16);
  return cat8(lo, hi);
}
// B-operand fragment: elements 0..7 at p, elements 8..15 at p+16
// (same K order as A; lane>=16 folds "+8" into p).
__device__ inline h16_t ldB16(const _Float16* p) {
  h8_t lo = *(const h8_t*)p;
  h8_t hi = *(const h8_t*)(p + 16);
  return cat8(lo, hi);
}
__device__ inline h16_t zero16() {
  h16_t z;
  #pragma unroll
  for (int i = 0; i < 16; ++i) z[i] = (_Float16)0.0f;
  return z;
}
__device__ inline f8_t wmma16(h16_t a, h16_t b, f8_t c) {
  return __builtin_amdgcn_wmma_f32_16x16x32_f16(false, a, false, b, (short)0, c,
                                                false, false);
}

// ---------------------------------------------------------------------------
// Tensor Data Mover: contiguous 1-D copy global -> LDS.
// D# per CDNA5 ISA ch.8: group0 {count=1, lds_addr, global_addr, type=2},
// group1 {data_size=8B, tensor_dim0=n, tensor_dim1=1, tile_dim0=n, stride=n}.
// 2-group form (tensors up to 2D; VADDR2/VADDR3 = NULL).  Issue from ONE wave
// per block (TDM ignores EXEC), then s_wait_tensorcnt + block barrier.
// ---------------------------------------------------------------------------
__device__ inline void tdm_copy_1d(const void* gsrc, unsigned lds_off,
                                   unsigned bytes) {
  unsigned n = bytes >> 3;                     // 8-byte elements, n <= 65535
  unsigned long long ga = (unsigned long long)gsrc;
  u32x4 g0;
  g0.x = 1u;                                   // count=1 (valid descriptor)
  g0.y = lds_off;                              // lds_addr [63:32]
  g0.z = (unsigned)ga;                         // global_addr [95:64]
  g0.w = (unsigned)(ga >> 32) | 0x80000000u;   // addr[56:32] | type=2 @ [127:126]
  u32x8 g1;
  g1.s0 = 3u << 16;                            // data_size=3 (8B), mask=0
  g1.s1 = (n & 0xffffu) << 16;                 // tensor_dim0[15:0]  @ bits[63:48]
  g1.s2 = (n >> 16) | (1u << 16);              // tensor_dim0[31:16] | tensor_dim1=1
  g1.s3 = n << 16;                             // tile_dim0 @ bits[127:112]
  g1.s4 = 0u;                                  // tile_dim1=0, tile_dim2=0
  g1.s5 = n;                                   // tensor_dim0_stride low 32
  g1.s6 = 0u;
  g1.s7 = 0u;
  asm volatile("tensor_load_to_lds %0, %1" :: "s"(g0), "s"(g1) : "memory");
}

// ---------------------------------------------------------------------------
// Prep kernels (weight conversion / layout transforms)
// ---------------------------------------------------------------------------

// conv weights -> f16, layout [layer][tap][outC][inC]
__global__ __launch_bounds__(256) void prep_convw(const float* __restrict__ res_w,
                                                  const float* __restrict__ tail_w,
                                                  _Float16* __restrict__ dst) {
  int id = blockIdx.x * 256 + threadIdx.x;
  if (id >= 33 * CONVW_PER) return;
  int l   = id / CONVW_PER;
  int r   = id - l * CONVW_PER;
  int tap = r >> 12;
  int o   = (r >> 6) & 63;
  int ic  = r & 63;
  int kh = tap / 3, kw = tap - 3 * kh;
  float v;
  if (l < 32) v = res_w[(((l * 64 + o) * 64 + ic) * 3 + kh) * 3 + kw];
  else        v = tail_w[((o * 64 + ic) * 3 + kh) * 3 + kw];
  dst[id] = (_Float16)v;
}

// MLP weight [K][256] f32 -> transposed [256][K] f16 (w1: first 576 rows only)
__global__ __launch_bounds__(256) void prep_wT(const float* __restrict__ src,
                                               _Float16* __restrict__ dst, int K) {
  int id = blockIdx.x * 256 + threadIdx.x;
  if (id >= 256 * K) return;
  int n = id / K, k = id - n * K;
  dst[id] = (_Float16)src[k * 256 + n];
}

// f32 [64][128][128] -> f16 [128][128][64]   (activation layout for conv B ops)
__global__ __launch_bounds__(256) void cvt_chw_hwc(const float* __restrict__ src,
                                                   _Float16* __restrict__ dst) {
  int id  = blockIdx.x * 256 + threadIdx.x;   // HW*64
  int pix = id >> 6, c = id & 63;
  dst[id] = (_Float16)src[c * HW + pix];
}

// Build unfolded feature tensor funf[pix][576], k = c*9 + kh*3 + kw, zero pad.
__global__ __launch_bounds__(256) void build_funf(const float* __restrict__ feat,
                                                  _Float16* __restrict__ funf) {
  int id  = blockIdx.x * 256 + threadIdx.x;   // HW*64
  int pix = id >> 6, c = id & 63;
  int y = pix >> 7, x = pix & 127;
  _Float16* o = funf + (size_t)pix * 576 + c * 9;
  const float* fp = feat + (size_t)c * HW;
  #pragma unroll
  for (int kh = 0; kh < 3; ++kh)
    #pragma unroll
    for (int kw = 0; kw < 3; ++kw) {
      int ys = y + kh - 1, xs = x + kw - 1;
      float v = (ys >= 0 && ys < 128 && xs >= 0 && xs < 128)
                    ? fp[ys * 128 + xs] : 0.0f;
      o[kh * 3 + kw] = (_Float16)v;
    }
}

// ---------------------------------------------------------------------------
// EDSR head: 3->64 conv3x3, tiny K so plain VALU
// ---------------------------------------------------------------------------
__global__ __launch_bounds__(256) void head_conv(const float* __restrict__ x,
                                                 const float* __restrict__ w,
                                                 const float* __restrict__ b,
                                                 float* __restrict__ out) {
  int id  = blockIdx.x * 256 + threadIdx.x;   // 64*HW
  int pix = id & (HW - 1);
  int oc  = id >> 14;
  int y = pix >> 7, xx = pix & 127;
  float acc = b[oc];
  #pragma unroll
  for (int c = 0; c < 3; ++c)
    #pragma unroll
    for (int kh = 0; kh < 3; ++kh) {
      int ys = y + kh - 1;
      if (ys < 0 || ys >= 128) continue;
      #pragma unroll
      for (int kw = 0; kw < 3; ++kw) {
        int xs = xx + kw - 1;
        if (xs < 0 || xs >= 128) continue;
        acc += x[c * HW + ys * 128 + xs] * w[((oc * 3 + c) * 3 + kh) * 3 + kw];
      }
    }
  out[id] = acc;
}

// ---------------------------------------------------------------------------
// 64->64 conv3x3 via WMMA implicit GEMM. M=outC tile(16), N=pixel tile(16),
// K=inC(64) x 9 taps. act: [y][x][c] f16 (global). Layer weights TDM-staged
// into LDS once per block; A-operands read via ds_load_b128.
// ---------------------------------------------------------------------------
__global__ __launch_bounds__(256) void conv64_wmma(
    const _Float16* __restrict__ act, const _Float16* __restrict__ wT,
    const float* __restrict__ bias, const float* __restrict__ addsrc,
    float* __restrict__ dst, int relu) {
  extern __shared__ char smemc[];
  _Float16* wlds = (_Float16*)smemc;           // 73728 B layer weights

  if ((threadIdx.x >> 5) == 0) {               // one wave issues the TDM op
    tdm_copy_1d(wT, 0u, CONVW_PER * 2u);
    __builtin_amdgcn_s_wait_tensorcnt(0);
  }
  __syncthreads();

  int wave = threadIdx.x >> 5;
  int lane = threadIdx.x & 31;
  int n16  = lane & 15;
  int hi   = lane >> 4;

  int tile = blockIdx.x * 8 + wave;   // 4096 tiles
  int pt = tile >> 2;                 // pixel tile 0..1023
  int ot = tile & 3;                  // outC tile 0..3
  int y  = pt >> 3;
  int x0 = (pt & 7) << 4;
  int oc0 = ot << 4;

  f8_t acc;
  #pragma unroll
  for (int i = 0; i < 8; ++i) acc[i] = 0.0f;

  #pragma unroll
  for (int tap = 0; tap < 9; ++tap) {
    int dy = tap / 3 - 1, dx = tap - 3 * (tap / 3) - 1;
    int ys = y + dy;
    if (ys < 0 || ys >= 128) continue;          // wave-uniform
    int xs  = x0 + n16 + dx;
    bool inx = (xs >= 0) && (xs < 128);
    int xsc = min(max(xs, 0), 127);             // clamped: always-legal load
    const _Float16* ab = act + (((ys << 7) + xsc) << 6);
    const _Float16* wb = wlds + (((tap << 6) + oc0 + n16) << 6);
    #pragma unroll
    for (int kc = 0; kc < 2; ++kc) {
      h16_t A = ldA16(wb + kc * 32 + hi * 8);
      h16_t B = ldB16(ab + kc * 32 + hi * 8);
      B = inx ? B : zero16();                   // VALU select, no exec branch
      acc = wmma16(A, B, acc);
    }
  }

  int xo = x0 + n16;
  #pragma unroll
  for (int v = 0; v < 8; ++v) {
    int oc = oc0 + v + hi * 8;
    float val = acc[v] + bias[oc];
    if (relu) val = fmaxf(val, 0.0f);
    size_t o = (size_t)oc * HW + (y << 7) + xo;
    if (addsrc) val += addsrc[o];
    dst[o] = val;
  }
}

// ---------------------------------------------------------------------------
// MLP 256x256 hidden layer: 16-point x 256-feature GEMM. A from per-wave LDS
// act buffers (stride 264 halves), B from the TDM-staged LDS weight tile.
// ---------------------------------------------------------------------------
__device__ __forceinline__ void gemm256(const _Float16* src,
                                        const _Float16* wT,
                                        const float* __restrict__ bias,
                                        _Float16* dst, int n16, int hiA, int hiB) {
  f8_t acc[16];
  #pragma unroll
  for (int t = 0; t < 16; ++t)
    #pragma unroll
    for (int i = 0; i < 8; ++i) acc[t][i] = 0.0f;

  const _Float16* arow = src + n16 * 264;
  #pragma unroll
  for (int kc = 0; kc < 8; ++kc) {
    h16_t A = ldA16(arow + kc * 32 + hiA);
    #pragma unroll
    for (int nt = 0; nt < 16; ++nt) {
      const _Float16* wp = wT + (size_t)(nt * 16 + n16) * 256 + kc * 32 + hiB;
      acc[nt] = wmma16(A, ldB16(wp), acc[nt]);
    }
  }
  #pragma unroll
  for (int nt = 0; nt < 16; ++nt) {
    int n = nt * 16 + n16;
    float bs = bias[n];
    #pragma unroll
    for (int v = 0; v < 8; ++v)
      dst[(v + hiA) * 264 + n] = (_Float16)fmaxf(acc[nt][v] + bs, 0.0f);
  }
}

// ---------------------------------------------------------------------------
// Fused LIIF query kernel: 4 shifted samples -> MLP -> area-weighted blend.
// One wave owns 16 query points; 4 waves / block; acts ping-pong in LDS;
// hidden-layer weights (w2/w3/w4) TDM-staged into a shared LDS tile.
// ---------------------------------------------------------------------------
__global__ __launch_bounds__(128, 1) void mlp_kernel(
    const _Float16* __restrict__ funf, const float* __restrict__ w1f,
    const _Float16* __restrict__ w1T, const _Float16* __restrict__ w2T,
    const _Float16* __restrict__ w3T, const _Float16* __restrict__ w4T,
    const float* __restrict__ b1, const float* __restrict__ b2,
    const float* __restrict__ b3, const float* __restrict__ b4,
    const float* __restrict__ w5, const float* __restrict__ b5,
    float* __restrict__ out) {
  extern __shared__ char smem[];
  int wave = threadIdx.x >> 5;
  int lane = threadIdx.x & 31;
  int n16  = lane & 15;
  int hi   = lane >> 4;
  int hiA  = hi * 8;    // A-operand / acc row offset
  int hiB  = hi * 8;    // B-operand K offset
  _Float16* buf0 = (_Float16*)(smem + wave * 16896);
  _Float16* buf1 = buf0 + 4224;                       // 16*264 halves each
  _Float16* wlds = (_Float16*)(smem + MLP_ACT_BYTES); // 128KB weight tile

  // Lanes L and L+16 own the same query point -> no broadcast needed for A.
  int p  = (blockIdx.x * 4 + wave) * 16 + n16;     // 2304*64 == 147456 exact
  int oy = p / 384, ox = p - oy * 384;
  float ga = -1.0f + (ox + 1) * (2.0f / 384.0f);   // grid[:,0] (reference flip)
  float gb = -1.0f + (oy + 1) * (2.0f / 384.0f);   // grid[:,1]

  float ar[4], qg0v[4], qg1v[4];
  int pixv[4];
  #pragma unroll
  for (int s = 0; s < 4; ++s) {
    float r = (s >> 1) ? 1.0f : -1.0f;
    float c = (s & 1) ? 1.0f : -1.0f;
    float fg0 = fminf(fmaxf(ga + r * (1.0f / 128.0f), -1.0f + 1e-6f), 1.0f - 1e-6f);
    float fg1 = fminf(fmaxf(gb + c * (1.0f / 128.0f), -1.0f + 1e-6f), 1.0f - 1e-6f);
    int iy = (int)rintf((fg0 + 1.0f) * 64.0f - 0.5f); iy = min(max(iy, 0), 127);
    int ix = (int)rintf((fg1 + 1.0f) * 64.0f - 0.5f); ix = min(max(ix, 0), 127);
    float qg0 = (ga - (-1.0f + (ix + 1) * (2.0f / 128.0f))) * 128.0f;
    float qg1 = (gb - (-1.0f + (iy + 1) * (2.0f / 128.0f))) * 128.0f;
    ar[s] = fabsf(qg0 * qg1) + 1e-9f;
    qg0v[s] = qg0; qg1v[s] = qg1; pixv[s] = iy * 128 + ix;
  }
  float total = ar[0] + ar[1] + ar[2] + ar[3];
  float r0 = 0.0f, r1 = 0.0f, r2 = 0.0f;

  for (int s = 0; s < 4; ++s) {
    float qg0 = qg0v[s], qg1 = qg1v[s];
    // q_grid of the point that accumulator row (v + hiA) belongs to
    float qA[8], qB[8];
    #pragma unroll
    for (int v = 0; v < 8; ++v) {
      qA[v] = __shfl(qg0, v + hiA, 32);
      qB[v] = __shfl(qg1, v + hiA, 32);
    }

    // ---- layer 1: [16 x 576] x [576 x 256]; w1T streamed from L2 (294KB,
    //      doesn't fit LDS). K rows 576..579 (q_grid, q_cell) folded in as a
    //      rank-1 VALU epilogue. ----
    const _Float16* ap = funf + (size_t)pixv[s] * 576;
    f8_t acc[16];
    #pragma unroll
    for (int t = 0; t < 16; ++t)
      #pragma unroll
      for (int i = 0; i < 8; ++i) acc[t][i] = 0.0f;

    for (int kc = 0; kc < 18; ++kc) {
      h16_t A = ldA16(ap + kc * 32 + hiA);
      if (kc + 1 < 18) __builtin_prefetch(ap + (kc + 1) * 32, 0, 3);
      #pragma unroll
      for (int nt = 0; nt < 16; ++nt) {
        const _Float16* wp = w1T + (size_t)(nt * 16 + n16) * 576 + kc * 32 + hiB;
        acc[nt] = wmma16(A, ldB16(wp), acc[nt]);
      }
    }
    #pragma unroll
    for (int nt = 0; nt < 16; ++nt) {
      int n = nt * 16 + n16;
      float w576 = w1f[576 * 256 + n];
      float w577 = w1f[577 * 256 + n];
      float cb = (2.0f / 3.0f) * (w1f[578 * 256 + n] + w1f[579 * 256 + n]) + b1[n];
      #pragma unroll
      for (int v = 0; v < 8; ++v) {
        float val = acc[nt][v] + qA[v] * w576 + qB[v] * w577 + cb;
        buf0[(v + hiA) * 264 + n] = (_Float16)fmaxf(val, 0.0f);
      }
    }

    // ---- layers 2..4: TDM-stage each 128KB weight matrix into LDS, then
    //      all 4 waves consume it ----
    __syncthreads();
    if (wave == 0) {
      tdm_copy_1d(w2T, MLP_ACT_BYTES, MLP_W_BYTES);
      __builtin_amdgcn_s_wait_tensorcnt(0);
    }
    __syncthreads();
    gemm256(buf0, wlds, b2, buf1, n16, hiA, hiB);

    __syncthreads();
    if (wave == 0) {
      tdm_copy_1d(w3T, MLP_ACT_BYTES, MLP_W_BYTES);
      __builtin_amdgcn_s_wait_tensorcnt(0);
    }
    __syncthreads();
    gemm256(buf1, wlds, b3, buf0, n16, hiA, hiB);

    __syncthreads();
    if (wave == 0) {
      tdm_copy_1d(w4T, MLP_ACT_BYTES, MLP_W_BYTES);
      __builtin_amdgcn_s_wait_tensorcnt(0);
    }
    __syncthreads();
    gemm256(buf0, wlds, b4, buf1, n16, hiA, hiB);

    // ---- layer 5 (256 -> 3) + diagonal-area weighting ----
    if (!hi) {
      float o0 = b5[0], o1 = b5[1], o2 = b5[2];
      const _Float16* arow = buf1 + n16 * 264;
      for (int k = 0; k < 256; ++k) {
        float av = (float)arow[k];
        o0 += av * w5[k * 3 + 0];
        o1 += av * w5[k * 3 + 1];
        o2 += av * w5[k * 3 + 2];
      }
      float wgt = ar[3 - s] / total;
      r0 += o0 * wgt; r1 += o1 * wgt; r2 += o2 * wgt;
    }
  }
  if (!hi) {
    size_t o = (size_t)p * 3;
    out[o] = r0; out[o + 1] = r1; out[o + 2] = r2;
  }
}

// ---------------------------------------------------------------------------
// Host launcher
// ---------------------------------------------------------------------------
extern "C" void kernel_launch(void* const* d_in, const int* in_sizes, int n_in,
                              void* d_out, int out_size, void* d_ws, size_t ws_size,
                              hipStream_t stream) {
  (void)in_sizes; (void)n_in; (void)out_size; (void)ws_size;
  const float* x      = (const float*)d_in[0];
  const float* head_w = (const float*)d_in[1];
  const float* head_b = (const float*)d_in[2];
  const float* res_w  = (const float*)d_in[3];
  const float* res_b  = (const float*)d_in[4];
  const float* tail_w = (const float*)d_in[5];
  const float* tail_b = (const float*)d_in[6];
  const float* w1 = (const float*)d_in[7];
  const float* b1 = (const float*)d_in[8];
  const float* w2 = (const float*)d_in[9];
  const float* b2 = (const float*)d_in[10];
  const float* w3 = (const float*)d_in[11];
  const float* b3 = (const float*)d_in[12];
  const float* w4 = (const float*)d_in[13];
  const float* b4 = (const float*)d_in[14];
  const float* w5 = (const float*)d_in[15];
  const float* b5 = (const float*)d_in[16];

  char* ws = (char*)d_ws;
  size_t off = 0;
  auto take = [&](size_t bytes) -> char* {
    char* p = ws + off;
    off = (off + bytes + 255) & ~(size_t)255;
    return p;
  };
  float*     featHead = (float*)take((size_t)64 * HW * 4);
  float*     hbuf     = (float*)take((size_t)64 * HW * 4);
  float*     tbuf     = (float*)take((size_t)64 * HW * 4);
  _Float16*  s16      = (_Float16*)take((size_t)HW * 64 * 2);
  _Float16*  funf     = (_Float16*)take((size_t)HW * 576 * 2);
  _Float16*  convw    = (_Float16*)take((size_t)33 * CONVW_PER * 2);
  _Float16*  w1T      = (_Float16*)take((size_t)256 * 576 * 2);
  _Float16*  w2T      = (_Float16*)take((size_t)256 * 256 * 2);
  _Float16*  w3T      = (_Float16*)take((size_t)256 * 256 * 2);
  _Float16*  w4T      = (_Float16*)take((size_t)256 * 256 * 2);

  // weight prep
  prep_convw<<<(33 * CONVW_PER + 255) / 256, 256, 0, stream>>>(res_w, tail_w, convw);
  prep_wT<<<(256 * 576 + 255) / 256, 256, 0, stream>>>(w1, w1T, 576);
  prep_wT<<<(256 * 256 + 255) / 256, 256, 0, stream>>>(w2, w2T, 256);
  prep_wT<<<(256 * 256 + 255) / 256, 256, 0, stream>>>(w3, w3T, 256);
  prep_wT<<<(256 * 256 + 255) / 256, 256, 0, stream>>>(w4, w4T, 256);

  // EDSR backbone
  head_conv<<<64 * HW / 256, 256, 0, stream>>>(x, head_w, head_b, featHead);
  const float* cur = featHead;
  for (int i = 0; i < 16; ++i) {
    cvt_chw_hwc<<<HW * 64 / 256, 256, 0, stream>>>(cur, s16);
    conv64_wmma<<<512, 256, CONVW_PER * 2, stream>>>(
        s16, convw + (size_t)(2 * i) * CONVW_PER,
        res_b + (2 * i) * 64, nullptr, tbuf, 1);
    cvt_chw_hwc<<<HW * 64 / 256, 256, 0, stream>>>(tbuf, s16);
    conv64_wmma<<<512, 256, CONVW_PER * 2, stream>>>(
        s16, convw + (size_t)(2 * i + 1) * CONVW_PER,
        res_b + (2 * i + 1) * 64, cur, hbuf, 0);
    cur = hbuf;
  }
  cvt_chw_hwc<<<HW * 64 / 256, 256, 0, stream>>>(hbuf, s16);
  conv64_wmma<<<512, 256, CONVW_PER * 2, stream>>>(
      s16, convw + (size_t)32 * CONVW_PER, tail_b, featHead, tbuf, 0); // f + h

  // unfold + fused LIIF MLP
  build_funf<<<HW * 64 / 256, 256, 0, stream>>>(tbuf, funf);
  mlp_kernel<<<2304, 128, MLP_ACT_BYTES + MLP_W_BYTES, stream>>>(
      funf, w1, w1T, w2T, w3T, w4T, b1, b2, b3, b4, w5, b5, (float*)d_out);
}
